// HarmonicFullyConnectedTensorProduct_26628797235374
// MI455X (gfx1250) — matrix operand order repaired and checked by reading, not compiled, not run on hardware
//
#include <hip/hip_runtime.h>

// ---------------------------------------------------------------------------
// HarmonicFullyConnectedTensorProduct, lmax=2, MUL=64, BATCH=1024, 11 paths.
// Fused single kernel: per (batch-tile of 16, l3-group) block, 4 waves.
//   Y[z,v,m,k] = sum_n C[m,n,k] x2[z,v,n]              (fp32, LDS, per path)
//   T_u[z,v,k] = sum_m x1[z,u,m] Y[z,v,m,k]            (v_pk_fma_f32 -> bf16 LDS)
//   out[z,w,k]+= sum_v W[p,w,u,v] T_u[z,v,k]           (v_wmma_f32_16x16x32_bf16)
// K-dim of WMMA = v (32/chunk) streamed over u, f32 accumulation.
// Double-buffered T/W staging; staging VALU co-executes with the matrix pipe.
// ---------------------------------------------------------------------------

typedef __attribute__((ext_vector_type(16))) __bf16 v16bf;
typedef __attribute__((ext_vector_type(8)))  float  v8f;
typedef __attribute__((ext_vector_type(2)))  float  v2f;

#define NPATHS 11

constexpr int P_L1[NPATHS] = {0,0,0,1,1,1,1,2,2,2,2};
constexpr int P_L2[NPATHS] = {0,1,2,0,1,1,2,0,1,2,2};
// paths grouped by l3 (output irrep): disjoint output slices -> no atomics
constexpr int G_NP[3]       = {3,4,4};
constexpr int G_PATHS[3][4] = {{0,4,9,0},{1,3,6,8},{2,5,7,10}};
constexpr int G_KOFS[3]     = {0,1,4};

struct Params {
  const float* x1[3];
  const float* x2[3];
  const float* W;
  const float* C[NPATHS];
  float* out;
};

// assemble a 16xbf16 WMMA fragment from two 16B-aligned LDS quads
static __device__ inline v16bf ld_frag(const unsigned short* p0, const unsigned short* p1) {
  union { unsigned int u[8]; v16bf b; } f;
  const uint4 q0 = *(const uint4*)p0;
  const uint4 q1 = *(const uint4*)p1;
  f.u[0]=q0.x; f.u[1]=q0.y; f.u[2]=q0.z; f.u[3]=q0.w;
  f.u[4]=q1.x; f.u[5]=q1.y; f.u[6]=q1.z; f.u[7]=q1.w;
  return f.b;
}

template<int G, int P>
static __device__ void process_path(const Params& pr, float* X1, float* X2, float* Ys,
                                    unsigned short* Ts, unsigned short* WB, float* CS,
                                    v8f* acc, int tid, int wave, int lane) {
  constexpr int L1 = P_L1[P];
  constexpr int L2 = P_L2[P];
  constexpr int L3 = (G == 0) ? 0 : ((G == 1) ? 1 : 2);
  constexpr int M1 = 2*L1 + 1, N2 = 2*L2 + 1, KK = 2*L3 + 1;
  constexpr int OFF1 = (L1 == 0) ? 0 : ((L1 == 1) ? 1024 : 4096);
  constexpr int OFF2 = (L2 == 0) ? 0 : ((L2 == 1) ? 1024 : 4096);
  constexpr int CN = M1 * N2 * KK;

  // CG tensor into LDS (tiny)
  if (tid < CN) CS[tid] = pr.C[P][tid];
  __syncthreads();

  // Y[(m*KK+k)*1024 + z*64 + v] = sum_n C[m,n,k] * x2[z,v,n]   (u-independent)
  // rolled m/k loops: no div/mod, C coeffs hoisted per (m,k), compact code.
  #pragma unroll 1
  for (int m = 0; m < M1; ++m) {
    #pragma unroll 1
    for (int k = 0; k < KK; ++k) {
      float cf[N2];
      #pragma unroll
      for (int n = 0; n < N2; ++n) cf[n] = CS[(m * N2 + n) * KK + k];
      float* yo = Ys + (m * KK + k) * 1024;
      #pragma unroll 1
      for (int zv = tid; zv < 1024; zv += 128) {
        float a = 0.f;
        #pragma unroll
        for (int n = 0; n < N2; ++n) a += cf[n] * X2[OFF2 + zv * N2 + n];
        yo[zv] = a;
      }
    }
  }
  __syncthreads();

  const int zt  = tid >> 3;          // T-compute: this thread's z row (0..15)
  const int v0  = (tid & 7) * 8;     // and v block (8 contiguous v)
  const int l15 = lane & 15;         // WMMA: A row (z) / B col (w) index
  const int kh  = lane >> 4;         // lane half selects K sub-block

  // Stateful staging bases: advance by one u per call -> no per-u address math.
  // W[p,w,u,v] flat: p*262144 + w*4096 + u*64 + v. Thread covers j=(tid+i*128)*4,
  // i.e. v = (tid*4)&63, w = (tid*4>>6) + 8*i  ->  i term = immediate offset.
  const float* wptr = pr.W + P * 262144
                    + ((tid * 4) >> 6) * 4096 + ((tid * 4) & 63);   // += 64 per u
  const float* x1p = X1 + OFF1 + zt * 64 * M1;                      // += M1 per u
  const float* yb  = Ys + zt * 64 + v0;          // Y loads: immediate offsets only

  auto stage = [&](unsigned short* Tb, unsigned short* Wb) {
    // ---- W[p,:,u,:] -> bf16 (w-major, v contiguous) ----
    unsigned short* wsb = Wb + tid * 4;
    #pragma unroll
    for (int i = 0; i < 8; ++i) {
      const float4 f = *(const float4*)(wptr + i * 32768);
      union { __bf16 h[4]; uint2 q; } pk;
      pk.h[0] = (__bf16)f.x; pk.h[1] = (__bf16)f.y;
      pk.h[2] = (__bf16)f.z; pk.h[3] = (__bf16)f.w;
      *(uint2*)(wsb + i * 512) = pk.q;
    }
    wptr += 64;
    // ---- T_u[k][z][v] = sum_m x1[z,u,m] * Y[m,k][z][v] -> bf16 ----
    float x1v[M1];
    #pragma unroll
    for (int m = 0; m < M1; ++m) x1v[m] = x1p[m];
    x1p += M1;
    unsigned short* tb = Tb + zt * 64 + v0;
    #pragma unroll
    for (int k = 0; k < KK; ++k) {
      v2f av[4];
      #pragma unroll
      for (int j = 0; j < 4; ++j) { v2f z0{}; av[j] = z0; }
      #pragma unroll
      for (int m = 0; m < M1; ++m) {
        union { float4 f4; v2f h[2]; } y0, y1;
        y0.f4 = *(const float4*)(yb + (m * KK + k) * 1024);
        y1.f4 = *(const float4*)(yb + (m * KK + k) * 1024 + 4);
        const v2f xm = {x1v[m], x1v[m]};
        av[0] += xm * y0.h[0];                 // v_pk_fma_f32
        av[1] += xm * y0.h[1];
        av[2] += xm * y1.h[0];
        av[3] += xm * y1.h[1];
      }
      union { __bf16 h[8]; uint4 q; } pk;
      #pragma unroll
      for (int j = 0; j < 4; ++j) {
        pk.h[2*j]   = (__bf16)av[j].x;         // v_cvt_pk_bf16_f32
        pk.h[2*j+1] = (__bf16)av[j].y;
      }
      *(uint4*)(tb + k * 1024) = pk.q;
    }
  };

  stage(Ts, WB);
  __syncthreads();

  #pragma unroll 1
  for (int u = 0; u < 64; ++u) {
    const unsigned short* Tc = Ts + (u & 1) * 5120;
    const unsigned short* Wc = WB + (u & 1) * 4096;

    // ---- fragments ----
    // B (K x 16, K=v chunk of 32): lane<16 holds K {0..7,16..23}, lane>=16 holds {8..15,24..31}
    v16bf bfrag[2];
    #pragma unroll
    for (int c = 0; c < 2; ++c) {
      const unsigned short* bp = Wc + (wave * 16 + l15) * 64 + c * 32 + kh * 8;
      bfrag[c] = ld_frag(bp, bp + 16);
    }
    // A (16 x 32): lane<16 holds K {0..7,16..23}, lane>=16 holds {8..15,24..31}
    v16bf afrag[KK][2];
    #pragma unroll
    for (int k = 0; k < KK; ++k) {
      #pragma unroll
      for (int c = 0; c < 2; ++c) {
        const unsigned short* tp = Tc + k * 1024 + l15 * 64 + c * 32;
        afrag[k][c] = ld_frag(tp + kh * 8, tp + 16 + kh * 8);
      }
    }
    // ---- WMMA burst (c outer: dependent acc[k] pairs spaced KK apart) ----
    #pragma unroll
    for (int c = 0; c < 2; ++c)
      #pragma unroll
      for (int k = 0; k < KK; ++k)
        acc[k] = __builtin_amdgcn_wmma_f32_16x16x32_bf16(
            false, afrag[k][c], false, bfrag[c], (short)0, acc[k], false, false);

    // ---- stage u+1 into the other buffer (co-executes with matrix pipe) ----
    if (u < 63)
      stage(Ts + ((u + 1) & 1) * 5120, WB + ((u + 1) & 1) * 4096);

    __syncthreads();
  }
}

template<int G, int PI>
static __device__ void run_paths(const Params& pr, float* X1, float* X2, float* Ys,
                                 unsigned short* Ts, unsigned short* WB, float* CS,
                                 v8f* acc, int tid, int wave, int lane) {
  if constexpr (PI < G_NP[G]) {
    process_path<G, G_PATHS[G][PI]>(pr, X1, X2, Ys, Ts, WB, CS, acc, tid, wave, lane);
    run_paths<G, PI + 1>(pr, X1, X2, Ys, Ts, WB, CS, acc, tid, wave, lane);
  }
}

template<int G>
static __device__ void run_group(const Params& pr, char* smem, int z0) {
  constexpr int L3 = (G == 0) ? 0 : ((G == 1) ? 1 : 2);
  constexpr int KK = 2 * L3 + 1;

  float* X1 = (float*)smem;                            // 16x64x(1+3+5) = 9216 f32
  float* X2 = X1 + 9216;                               // 9216 f32
  float* Ys = X2 + 9216;                               // up to 25600 f32
  unsigned short* Ts = (unsigned short*)(Ys + 25600);  // 2 x 5*1024 bf16
  unsigned short* WB = Ts + 2 * 5120;                  // 2 x 64*64 bf16
  float* CS = (float*)(WB + 2 * 4096);                 // 128 f32

  const int tid = threadIdx.x;
  const int lane = tid & 31, wave = tid >> 5;

  // load this block's x1/x2 batch slice (all l) into LDS
  #pragma unroll
  for (int l = 0; l < 3; ++l) {
    const int ml = 2 * l + 1;
    const int off = (l == 0) ? 0 : ((l == 1) ? 1024 : 4096);
    const float* s1 = pr.x1[l] + z0 * 64 * ml;
    const float* s2 = pr.x2[l] + z0 * 64 * ml;
    #pragma unroll 1
    for (int i = tid; i < 1024 * ml; i += 128) {
      X1[off + i] = s1[i];
      X2[off + i] = s2[i];
    }
  }
  __syncthreads();

  v8f acc[KK];
  #pragma unroll
  for (int k = 0; k < KK; ++k) { v8f z{}; acc[k] = z; }

  run_paths<G, 0>(pr, X1, X2, Ys, Ts, WB, CS, acc, tid, wave, lane);

  // C/D layout: VGPR r, lane<16 -> (M=r, N=lane); lane>=16 -> (M=r+8, N=lane-16)
  #pragma unroll
  for (int k = 0; k < KK; ++k) {
    #pragma unroll
    for (int r = 0; r < 8; ++r) {
      const int zz = z0 + r + (lane >> 4) * 8;
      const int ww = wave * 16 + (lane & 15);
      pr.out[(zz * 64 + ww) * 9 + G_KOFS[G] + k] = acc[k][r];
    }
  }
}

__global__ void __launch_bounds__(128)
tp_kernel(Params pr) {
  extern __shared__ char smem[];
  const int z0 = blockIdx.x * 16;
  switch (blockIdx.y) {
    case 0: run_group<0>(pr, smem, z0); break;
    case 1: run_group<1>(pr, smem, z0); break;
    default: run_group<2>(pr, smem, z0); break;
  }
}

extern "C" void kernel_launch(void* const* d_in, const int* in_sizes, int n_in,
                              void* d_out, int out_size, void* d_ws, size_t ws_size,
                              hipStream_t stream) {
  (void)in_sizes; (void)n_in; (void)d_ws; (void)ws_size; (void)out_size;
  Params pr;
  pr.x1[0] = (const float*)d_in[0];
  pr.x1[1] = (const float*)d_in[1];
  pr.x1[2] = (const float*)d_in[2];
  pr.x2[0] = (const float*)d_in[3];
  pr.x2[1] = (const float*)d_in[4];
  pr.x2[2] = (const float*)d_in[5];
  pr.W = (const float*)d_in[6];
  for (int p = 0; p < NPATHS; ++p) pr.C[p] = (const float*)d_in[7 + p];
  pr.out = (float*)d_out;

  const size_t shmem = (9216 + 9216 + 25600) * sizeof(float)
                     + (2 * 5120 + 2 * 4096) * sizeof(unsigned short)
                     + 128 * sizeof(float);   // 213504 B, under 320 KB/WGP
  hipFuncSetAttribute((const void*)tp_kernel,
                      hipFuncAttributeMaxDynamicSharedMemorySize, (int)shmem);
  tp_kernel<<<dim3(64, 3, 1), dim3(128, 1, 1), shmem, stream>>>(pr);
}
